// GCN_1090921693297
// MI455X (gfx1250) — matrix operand order repaired and checked by reading, not compiled, not run on hardware
//
#include <hip/hip_runtime.h>
#include <hip/hip_bf16.h>
#include <math.h>

// ---------------------------------------------------------------------------
// GCN forward on MI455X (gfx1250, wave32):
//   h0   = x @ W1                      (bf16 WMMA, f32 accum)
//   hagg = scatter_add(edges, h0)      (global_atomic_add_f32, L2-resident)
//   g0   = relu(hagg) @ W2             (bf16 WMMA, relu fused into A-load)
//   lagg = scatter_add(edges, g0)
//   out  = log_softmax(lagg, axis=1)   (one wave per row)
// ---------------------------------------------------------------------------

typedef __attribute__((ext_vector_type(16))) __bf16 v16bf;
typedef __attribute__((ext_vector_type(8)))  __bf16 v8bf;
typedef __attribute__((ext_vector_type(8)))  float  v8f;

// Column stride (in bf16 elements) for the LDS B-tile. 40 elems = 80 bytes
// = 20 banks: lanes 0..15 hit banks (l*20 mod 64), all distinct -> no
// conflicts on the 16-byte fragment reads.
#define BCOL_STRIDE 40

// ---------------------------------------------------------------------------
// Tiled WMMA GEMM: C[M,N] = A[M,K] * B[K,N], fp32 in memory, bf16 operands,
// fp32 accumulation via v_wmma_f32_16x16x32_bf16.
// Block = 128 threads (4 waves); each wave computes a 16-row strip across
// all NTILES*16 columns; block covers 64 rows.
// B k-slab is staged in LDS **in fragment layout** Bs[column][k] so each
// lane's B fragment is two contiguous ds_load_b128's.
// ---------------------------------------------------------------------------
template<int NTILES, bool RELU_IN>
__global__ __launch_bounds__(128)
void wmma_gemm_bf16(const float* __restrict__ A, const float* __restrict__ B,
                    float* __restrict__ C, int M, int K, int N)
{
    constexpr int NPAD = NTILES * 16;
    __shared__ __align__(16) __bf16 Bs[NPAD * BCOL_STRIDE];

    const int lane = threadIdx.x & 31;
    const int wave = threadIdx.x >> 5;
    const int half = lane >> 4;               // which 16-lane half of the wave
    const int r    = lane & 15;

    const int row0 = blockIdx.x * 64 + wave * 16;     // wave's 16-row strip
    const int arow = min(row0 + r, M - 1);            // clamp for safe loads
    const float* __restrict__ Ar = A + (size_t)arow * K;

    v8f acc[NTILES];
#pragma unroll
    for (int t = 0; t < NTILES; ++t)
        acc[t] = (v8f){0.f,0.f,0.f,0.f,0.f,0.f,0.f,0.f};

    const int kb = half * 8;    // B frag: lanes 0-15 -> K {0..7,16..23}, 16-31 -> {8..15,24..31}

    for (int k0 = 0; k0 < K; k0 += 32) {
        // --- A fragment first: global loads issue before the LDS barrier. ---
        // ISA 7.12.2: lane(0-15): K = {k0+0..7, k0+16..23};
        //             lane(16-31): K = {k0+8..15, k0+24..31}
        // -> two contiguous 8-float runs -> 4x global float4 loads.
        v16bf a;
        {
            const float* p = Ar + k0 + half * 8;
            float4 q0 = *(const float4*)(p);
            float4 q1 = *(const float4*)(p + 4);
            float4 q2 = *(const float4*)(p + 16);
            float4 q3 = *(const float4*)(p + 20);
            float vv[16] = {q0.x,q0.y,q0.z,q0.w, q1.x,q1.y,q1.z,q1.w,
                            q2.x,q2.y,q2.z,q2.w, q3.x,q3.y,q3.z,q3.w};
#pragma unroll
            for (int i = 0; i < 16; ++i) {
                float f = vv[i];
                if (RELU_IN) f = fmaxf(f, 0.f);
                a[i] = (__bf16)f;
            }
        }

        // --- cooperative B staging: B[k0+kk, nn] -> Bs[nn][kk] (bf16) ---
        // Global reads coalesced over nn; LDS writes land in fragment layout.
        for (int idx = threadIdx.x; idx < 32 * NPAD; idx += 128) {
            const int kk = idx / NPAD;
            const int nn = idx - kk * NPAD;
            float v = (nn < N) ? B[(size_t)(k0 + kk) * N + nn] : 0.f;
            Bs[nn * BCOL_STRIDE + kk] = (__bf16)v;
        }
        __syncthreads();

        // --- NTILES back-to-back WMMAs; B frag = 2x ds_load_b128 each ---
#pragma unroll
        for (int t = 0; t < NTILES; ++t) {
            const __bf16* bcol = &Bs[(t * 16 + r) * BCOL_STRIDE + kb];
            v8bf lo = *(const v8bf*)(bcol);       // 16B aligned
            v8bf hi = *(const v8bf*)(bcol + 16);  // 16B aligned
            v16bf b = __builtin_shufflevector(lo, hi,
                0,1,2,3,4,5,6,7,8,9,10,11,12,13,14,15);
            acc[t] = __builtin_amdgcn_wmma_f32_16x16x32_bf16(
                false, a, false, b, (short)0, acc[t], false, false);
        }
        __syncthreads();
    }

    // --- store: D layout = col r, row row0 + half*8 + j (vgpr j) ---
#pragma unroll
    for (int t = 0; t < NTILES; ++t) {
        const int col = t * 16 + r;
        if (col >= N) continue;
#pragma unroll
        for (int j = 0; j < 8; ++j) {
            const int rr = row0 + half * 8 + j;
            if (rr < M) C[(size_t)rr * N + col] = acc[t][j];
        }
    }
}

// ---------------------------------------------------------------------------
// Edge scatter: out[dst] += w * feat[src]; one thread per (edge, float4 chunk)
// ---------------------------------------------------------------------------
__global__ __launch_bounds__(256)
void spmm_scatter_f4(const int* __restrict__ ei, const float* __restrict__ ew,
                     const float* __restrict__ feat, float* __restrict__ out,
                     int E, int F)
{
    const int chunks = F >> 2;
    const long long idx = (long long)blockIdx.x * blockDim.x + threadIdx.x;
    const long long total = (long long)E * chunks;
    if (idx >= total) return;
    const int e = (int)(idx / chunks);
    const int c = (int)(idx - (long long)e * chunks);
    const int s = ei[e];
    const int d = ei[E + e];
    const float w = ew[e];
    const float4 v = ((const float4*)(feat + (size_t)s * F))[c];
    float* o = out + (size_t)d * F + (c << 2);
    atomicAdd(o + 0, w * v.x);
    atomicAdd(o + 1, w * v.y);
    atomicAdd(o + 2, w * v.z);
    atomicAdd(o + 3, w * v.w);
}

// ---------------------------------------------------------------------------
// Zero-fill (accumulation buffers must be re-zeroed every call)
// ---------------------------------------------------------------------------
__global__ __launch_bounds__(256)
void zero_f32(float* __restrict__ p, long long n)
{
    const long long i = (long long)blockIdx.x * blockDim.x + threadIdx.x;
    if (i < n) p[i] = 0.f;
}

// ---------------------------------------------------------------------------
// Row-wise log_softmax, C<=64 classes: one wave32 per row, 8 rows per block
// ---------------------------------------------------------------------------
__global__ __launch_bounds__(256)
void logsoftmax_rows(const float* __restrict__ in, float* __restrict__ out,
                     int M, int C)
{
    const int wave = threadIdx.x >> 5, lane = threadIdx.x & 31;
    const int row = blockIdx.x * 8 + wave;
    if (row >= M) return;
    const float* r = in + (size_t)row * C;
    float v0 = (lane < C)      ? r[lane]      : -INFINITY;
    float v1 = (lane + 32 < C) ? r[lane + 32] : -INFINITY;
    float m = fmaxf(v0, v1);
#pragma unroll
    for (int off = 16; off > 0; off >>= 1)
        m = fmaxf(m, __shfl_xor(m, off, 32));
    float s = 0.f;
    if (lane < C)      s += expf(v0 - m);
    if (lane + 32 < C) s += expf(v1 - m);
#pragma unroll
    for (int off = 16; off > 0; off >>= 1)
        s += __shfl_xor(s, off, 32);
    const float lse = m + logf(s);
    if (lane < C)      out[(size_t)row * C + lane]      = v0 - lse;
    if (lane + 32 < C) out[(size_t)row * C + lane + 32] = v1 - lse;
}

// ---------------------------------------------------------------------------
extern "C" void kernel_launch(void* const* d_in, const int* in_sizes, int n_in,
                              void* d_out, int out_size, void* d_ws, size_t ws_size,
                              hipStream_t stream)
{
    const float* x  = (const float*)d_in[0];   // [100000, 512]
    const int*   ei = (const int*)  d_in[1];   // [2, E]
    const float* ew = (const float*)d_in[2];   // [E]
    const float* W1 = (const float*)d_in[3];   // [512, 256]
    const float* W2 = (const float*)d_in[4];   // [256, 40]

    const int M = 100000, F = 512, H = 256, C = 40;
    const int E = in_sizes[2];

    // workspace layout (~237 MB):
    float* h0   = (float*)d_ws;                 // M*H : x@W1
    float* hagg = h0   + (size_t)M * H;         // M*H : scatter target (zeroed)
    float* g0   = hagg + (size_t)M * H;         // M*C : relu(hagg)@W2
    float* lagg = g0   + (size_t)M * C;         // M*C : scatter target (zeroed)

    {   // zero hagg..lagg (contiguous; includes g0, harmless — overwritten)
        const long long n = (long long)M * H + 2LL * M * C;
        const int blocks = (int)((n + 255) / 256);
        zero_f32<<<blocks, 256, 0, stream>>>(hagg, n);
    }
    {   // GEMM1: h0 = x @ W1   (M x 512 x 256)
        const int blocks = (M + 63) / 64;
        wmma_gemm_bf16<16, false><<<blocks, 128, 0, stream>>>(x, W1, h0, M, F, H);
    }
    {   // SpMM1: hagg[dst] += w * h0[src]
        const long long total = (long long)E * (H / 4);
        const int blocks = (int)((total + 255) / 256);
        spmm_scatter_f4<<<blocks, 256, 0, stream>>>(ei, ew, h0, hagg, E, H);
    }
    {   // GEMM2: g0 = relu(hagg) @ W2   (M x 256 x 40, relu fused into A-load)
        const int blocks = (M + 63) / 64;
        wmma_gemm_bf16<3, true><<<blocks, 128, 0, stream>>>(hagg, W2, g0, M, H, C);
    }
    {   // SpMM2: lagg[dst] += w * g0[src]
        const long long total = (long long)E * (C / 4);
        const int blocks = (int)((total + 255) / 256);
        spmm_scatter_f4<<<blocks, 256, 0, stream>>>(ei, ew, g0, lagg, E, C);
    }
    {   // log_softmax -> d_out
        const int blocks = (M + 7) / 8;
        logsoftmax_rows<<<blocks, 256, 0, stream>>>(lagg, (float*)d_out, M, C);
    }
}
